// Spidercnn_seg_feature_35742717837597
// MI455X (gfx1250) — matrix operand order repaired and checked by reading, not compiled, not run on hardware
//
#include <hip/hip_runtime.h>
#include <hip/hip_bf16.h>

typedef __attribute__((ext_vector_type(16))) _Float16 v16h;
typedef __attribute__((ext_vector_type(8)))  _Float16 v8h;
typedef __attribute__((ext_vector_type(8)))  float    v8f;

#define NB   4
#define NPTS 4096
#define KNN  16

// ---------------------------------------------------------------------------
// 1) pc (B,N,6) -> feat0 (B,6,N)
// ---------------------------------------------------------------------------
__global__ void transpose_kernel(const float* __restrict__ pc, float* __restrict__ out) {
    int e = blockIdx.x * 256 + threadIdx.x;            // < B*6*4096
    if (e >= NB * 6 * NPTS) return;
    int b = e / (6 * NPTS);
    int rem = e - b * 6 * NPTS;
    int c = rem / NPTS;
    int n = rem - c * NPTS;
    out[(size_t)(b * 6 + c) * NPTS + n] = pc[((size_t)b * NPTS + n) * 6 + c];
}

// ---------------------------------------------------------------------------
// 2) KNN: for each point, 16 nearest (by squared distance) among same batch.
//    One thread per point; candidates tiled through LDS.
// ---------------------------------------------------------------------------
__global__ void knn_kernel(const float* __restrict__ pc, int* __restrict__ idxg) {
    __shared__ float cx[256], cy[256], cz[256];
    int b = blockIdx.x >> 4;                       // 16 blocks per batch
    int n = (blockIdx.x & 15) * 256 + threadIdx.x;
    const float* base = pc + (size_t)b * NPTS * 6;
    float x = base[n * 6 + 0], y = base[n * 6 + 1], z = base[n * 6 + 2];

    float dist[KNN];
    int   ind[KNN];
#pragma unroll
    for (int j = 0; j < KNN; ++j) { dist[j] = 3.0e38f; ind[j] = 0; }

    for (int m0 = 0; m0 < NPTS; m0 += 256) {
        __syncthreads();
        int mload = m0 + threadIdx.x;
        cx[threadIdx.x] = base[mload * 6 + 0];
        cy[threadIdx.x] = base[mload * 6 + 1];
        cz[threadIdx.x] = base[mload * 6 + 2];
        __syncthreads();
        for (int i = 0; i < 256; ++i) {
            float dx = x - cx[i], dy = y - cy[i], dz = z - cz[i];
            float d = dx * dx + dy * dy + dz * dz;
            int m = m0 + i;
            if (d < dist[KNN - 1]) {
                // fully unrolled branchless insertion (registers only)
#pragma unroll
                for (int t = KNN - 1; t >= 1; --t) {
                    bool gt   = d < dist[t - 1];
                    bool here = (d < dist[t]) && !gt;
                    if (gt)   { dist[t] = dist[t - 1]; ind[t] = ind[t - 1]; }
                    if (here) { dist[t] = d;           ind[t] = m; }
                }
                if (d < dist[0]) { dist[0] = d; ind[0] = m; }
            }
        }
    }
#pragma unroll
    for (int j = 0; j < KNN; ++j)
        idxg[((size_t)b * NPTS + n) * KNN + j] = ind[j];
}

// ---------------------------------------------------------------------------
// 3) Taylor basis -> per-layer 3-channel projection, stored f16.
//    tayh layout: [layer][b][t(3)][n][k]  (n*k stride 65536 per (b,t))
// ---------------------------------------------------------------------------
__global__ void tay_kernel(const float* __restrict__ pc, const int* __restrict__ idxg,
                           const float* w1_1, const float* b1_1,
                           const float* w1_2, const float* b1_2,
                           const float* w1_3, const float* b1_3,
                           const float* w1_4, const float* b1_4,
                           _Float16* __restrict__ tayh) {
    int e = blockIdx.x * 256 + threadIdx.x;        // < B*N*K
    if (e >= NB * NPTS * KNN) return;
    int b = e >> 16;                               // N*K = 65536
    int rem = e & 65535;
    int n = rem >> 4;
    const float* base = pc + (size_t)b * NPTS * 6;
    int m = idxg[e];
    float x = base[m * 6 + 0] - base[n * 6 + 0];
    float y = base[m * 6 + 1] - base[n * 6 + 1];
    float z = base[m * 6 + 2] - base[n * 6 + 2];
    float mono[20] = {1.0f, x, y, z,
                      x * x, x * y, x * z, y * y, y * z, z * z,
                      x * x * x, x * x * y, x * x * z, x * y * y, x * y * z,
                      x * z * z, y * y * y, y * y * z, y * z * z, z * z * z};
    const float* w1s[4] = {w1_1, w1_2, w1_3, w1_4};
    const float* b1s[4] = {b1_1, b1_2, b1_3, b1_4};
#pragma unroll
    for (int l = 0; l < 4; ++l) {
#pragma unroll
        for (int t = 0; t < 3; ++t) {
            float s = b1s[l][t];
#pragma unroll
            for (int i = 0; i < 20; ++i) s += w1s[l][t * 20 + i] * mono[i];
            tayh[((size_t)(l * NB + b) * 3 + t) * 65536 + rem] = (_Float16)s;
        }
    }
}

// ---------------------------------------------------------------------------
// 4) f32 -> f16 weight conversion
// ---------------------------------------------------------------------------
__global__ void cvt_kernel(const float* __restrict__ src, _Float16* __restrict__ dst, int n) {
    int e = blockIdx.x * 256 + threadIdx.x;
    if (e < n) dst[e] = (_Float16)src[e];
}

// ---------------------------------------------------------------------------
// 5) WMMA layer kernel.
//    GEMM: out[oc, pt] = sum_{c3,k} (gf[c3/3][pt][k]*tay[c3%3][pt][k]) * w2h[oc][c3*16+k]
//    Tile: 16 points per workgroup; gathered features staged in LDS (f16).
//    Each wave owns 16x16 output tiles; K consumed 32 at a time via
//    v_wmma_f32_16x16x32_f16, A and B both in contiguous k = hi*16 + j order.
// ---------------------------------------------------------------------------
__global__ void layer_kernel(const float* __restrict__ feat_in, float* __restrict__ feat_out,
                             const int* __restrict__ idxg, const _Float16* __restrict__ tayh_l,
                             const _Float16* __restrict__ w2h, const float* __restrict__ b2,
                             float* __restrict__ pf_out,
                             int cin, int cout, int ch_off) {
    extern __shared__ char smem[];
    _Float16* gf = (_Float16*)smem;            // [cin][16 pts][16 k]
    _Float16* tl = gf + cin * 256;             // [3][16 pts][16 k]

    const int tile = blockIdx.x;               // 1024 tiles of 16 points
    const int tid  = threadIdx.x;

    // cooperative gather of neighbor features into LDS (f16)
    for (int e = tid; e < cin * 256; e += 256) {
        int c = e >> 8, rem = e & 255, p = rem >> 4, k = rem & 15;
        int g = tile * 16 + p;
        int b = g >> 12, n = g & (NPTS - 1);
        int m = idxg[g * KNN + k];
        gf[e] = (_Float16)feat_in[((size_t)(b * cin + c)) * NPTS + m];
    }
    for (int e = tid; e < 768; e += 256) {
        int t = e >> 8, rem = e & 255, p = rem >> 4, k = rem & 15;
        int g = tile * 16 + p;
        int b = g >> 12, n = g & (NPTS - 1);
        tl[e] = tayh_l[((size_t)(b * 3 + t)) * 65536 + n * 16 + k];
    }
    __syncthreads();

    const int lane = tid & 31, wave = tid >> 5;
    const int hi = lane >> 4, lp = lane & 15;
    const int C3 = cin * 3, K3 = C3 * 16, chunks = C3 >> 1;
    const int mtiles = cout >> 4;

    for (int mt = wave; mt < mtiles; mt += 8) {
        v8f acc = {};
        const _Float16* aptr = w2h + (size_t)(mt * 16 + lp) * K3;
        for (int ch = 0; ch < chunks; ++ch) {
            const int kbase = ch * 32;
            // A fragment: row M=lane%16; elems 0..15 -> K=kbase+hi*16+j
            v8h a0 = *(const v8h*)(aptr + kbase + hi * 16);
            v8h a1 = *(const v8h*)(aptr + kbase + hi * 16 + 8);
            // B fragment: col N=lane%16; K = kbase + hi*16 + j  ->  c3 = 2*ch+hi, k=j
            int c3 = 2 * ch + hi;
            int ci = c3 / 3;
            int tt = c3 - ci * 3;
            const _Float16* gr = gf + ci * 256 + lp * 16;
            const _Float16* tr = tl + tt * 256 + lp * 16;
            v8h g0 = *(const v8h*)gr;       v8h g1 = *(const v8h*)(gr + 8);
            v8h t0 = *(const v8h*)tr;       v8h t1 = *(const v8h*)(tr + 8);
            v8h f0 = g0 * t0;               v8h f1 = g1 * t1;   // v_pk_mul_f16
            v16h af, bf;
#pragma unroll
            for (int i = 0; i < 8; ++i) {
                af[i] = a0[i]; af[i + 8] = a1[i];
                bf[i] = f0[i]; bf[i + 8] = f1[i];
            }
            acc = __builtin_amdgcn_wmma_f32_16x16x32_f16(
                false, af, false, bf, (short)0, acc, false, false);
        }
        // C/D layout: VGPR r -> M = r + 8*hi ; N = lane%16
        int g = tile * 16 + lp;
        int b = g >> 12, n = g & (NPTS - 1);
#pragma unroll
        for (int r = 0; r < 8; ++r) {
            int oc = mt * 16 + r + hi * 8;
            float v = acc[r] + b2[oc];
            v = v > 0.0f ? v : 0.0f;                               // ReLU
            feat_out[((size_t)(b * cout + oc)) * NPTS + n] = v;    // next-layer input
            pf_out[((size_t)(b * 480 + ch_off + oc)) * NPTS + n] = v; // point_feat slice
        }
    }
}

// ---------------------------------------------------------------------------
// 6) top-2 over the N axis for each (b, channel) row of point_feat
// ---------------------------------------------------------------------------
__global__ void top2_kernel(const float* __restrict__ pf, float* __restrict__ catf) {
    __shared__ float s1[256], s2[256];
    int row = blockIdx.x;                       // b*480 + ch
    const float* p = pf + (size_t)row * NPTS;
    float m1 = -3.0e38f, m2 = -3.0e38f;
    for (int i = threadIdx.x; i < NPTS; i += 256) {
        float v = p[i];
        if (v > m1) { m2 = m1; m1 = v; }
        else if (v > m2) { m2 = v; }
    }
    s1[threadIdx.x] = m1; s2[threadIdx.x] = m2;
    __syncthreads();
    for (int s = 128; s > 0; s >>= 1) {
        if (threadIdx.x < s) {
            float a1 = s1[threadIdx.x],     a2 = s2[threadIdx.x];
            float c1 = s1[threadIdx.x + s], c2 = s2[threadIdx.x + s];
            s1[threadIdx.x] = fmaxf(a1, c1);
            s2[threadIdx.x] = fmaxf(fminf(a1, c1), fmaxf(a2, c2));
        }
        __syncthreads();
    }
    if (threadIdx.x == 0) {
        int b = row / 480, ch = row - b * 480;
        catf[b * 960 + ch * 2 + 0] = s1[0];
        catf[b * 960 + ch * 2 + 1] = s2[0];
    }
}

// ---------------------------------------------------------------------------
extern "C" void kernel_launch(void* const* d_in, const int* in_sizes, int n_in,
                              void* d_out, int out_size, void* d_ws, size_t ws_size,
                              hipStream_t stream) {
    const float* pc = (const float*)d_in[0];
    const float* w1[4], *b1[4], *w2[4], *b2[4];
    for (int l = 0; l < 4; ++l) {
        w1[l] = (const float*)d_in[1 + 4 * l];
        b1[l] = (const float*)d_in[2 + 4 * l];
        w2[l] = (const float*)d_in[3 + 4 * l];
        b2[l] = (const float*)d_in[4 + 4 * l];
    }
    float* out  = (float*)d_out;
    float* catf = out;              // (B, 960)
    float* pf   = out + NB * 960;   // (B, 480, N)

    // workspace carve-up
    char* ws = (char*)d_ws;
    int*       idxg  = (int*)ws;                               ws += (size_t)NB * NPTS * KNN * 4;   // 1 MB
    _Float16*  tayh  = (_Float16*)ws;                          ws += (size_t)4 * NB * 3 * 65536 * 2; // 6 MB
    float*     featA = (float*)ws;                             ws += (size_t)NB * 256 * NPTS * 4;   // 16 MB
    float*     featB = (float*)ws;                             ws += (size_t)NB * 256 * NPTS * 4;   // 16 MB
    const int  cins[4]  = {6, 32, 64, 128};
    const int  couts[4] = {32, 64, 128, 256};
    const int  choff[4] = {0, 32, 96, 224};
    _Float16*  w2h[4];
    for (int l = 0; l < 4; ++l) {
        w2h[l] = (_Float16*)ws;
        ws += (size_t)couts[l] * cins[l] * 3 * KNN * 2;
    }

    // 1) transpose pc -> featA (B,6,N)
    transpose_kernel<<<(NB * 6 * NPTS + 255) / 256, 256, 0, stream>>>(pc, featA);
    // 2) KNN
    knn_kernel<<<NB * (NPTS / 256), 256, 0, stream>>>(pc, idxg);
    // 3) Taylor projections (f16)
    tay_kernel<<<(NB * NPTS * KNN + 255) / 256, 256, 0, stream>>>(
        pc, idxg, w1[0], b1[0], w1[1], b1[1], w1[2], b1[2], w1[3], b1[3], tayh);
    // 4) weight conversion f32 -> f16
    for (int l = 0; l < 4; ++l) {
        int n = couts[l] * cins[l] * 3 * KNN;
        cvt_kernel<<<(n + 255) / 256, 256, 0, stream>>>(w2[l], w2h[l], n);
    }
    // 5) WMMA layers, ping-pong feature buffers
    float* fin  = featA;
    float* fout = featB;
    const int ntiles = NB * NPTS / 16;   // 1024
    for (int l = 0; l < 4; ++l) {
        size_t shmem = (size_t)cins[l] * 256 * 2 + 768 * 2;
        layer_kernel<<<ntiles, 256, shmem, stream>>>(
            fin, fout, idxg, tayh + (size_t)l * NB * 3 * 65536,
            w2h[l], b2[l], pf, cins[l], couts[l], choff[l]);
        float* t = fin; fin = fout; fout = t;
    }
    // 6) top-2 reduction -> cat_feat
    top2_kernel<<<NB * 480, 256, 0, stream>>>(pf, catf);
}
